// PointNetPP_66168266162372
// MI455X (gfx1250) — matrix operand order repaired and checked by reading, not compiled, not run on hardware
//
#include <hip/hip_runtime.h>

// ---------------------------------------------------------------------------
// CDNA5 (gfx1250) PointNet++ forward.  wave32, WMMA f32<-f16 16x16x32 tiles.
// B operands pre-swizzled in LDS so each lane's fragment is one contiguous
// 32-byte run (2x ds_load_b128); A operands row-major (K-pairs coalesce to
// 2x ds_load_b128 per lane).
// ---------------------------------------------------------------------------

typedef __attribute__((ext_vector_type(16))) _Float16 v16h;
typedef __attribute__((ext_vector_type(8)))  float    v8f;

#define DEV_INLINE __device__ __forceinline__

DEV_INLINE v8f wmma32(v16h a, v16h b, v8f c) {
  // D(16x16,f32) = A(16x32,f16) * B(32x16,f16) + C
  return __builtin_amdgcn_wmma_f32_16x16x32_f16(
      /*neg_a=*/false, a, /*neg_b=*/false, b,
      /*c_mod=*/(short)0, c, /*reuse_a=*/false, /*reuse_b=*/false);
}

// A fragment: 16x32 f16 tile, row-major in LDS with leading dim `ld` (halves).
// lanes 0-15 -> M=lane; element (v,h): K = (lane>=16?16:0) + v*2 + h
DEV_INLINE v16h frag_A(const _Float16* A, int ld) {
  int lane = (int)(threadIdx.x & 31u);
  int g = lane >> 4, m = lane & 15;
  v16h a;
#pragma unroll
  for (int v = 0; v < 8; ++v) {
    int k = g * 16 + v * 2;
    a[2 * v]     = A[m * ld + k];
    a[2 * v + 1] = A[m * ld + k + 1];
  }
  return a;
}

// Swizzled-B storage index: element (k,n) of a 32x16 (KxN) tile lands at the
// owning lane's contiguous 16-half run:  lane = (k&16)|n, elem = k&15.
DEV_INLINE int swzB(int k, int n) { return ((k & 16) | n) * 16 + (k & 15); }

DEV_INLINE v16h frag_B_swz(const _Float16* base) {
  return *(const v16h*)(base + (threadIdx.x & 31u) * 16);
}

// ---------------------------------------------------------------------------
// lin_in:  h = relu(relu(x @ W0 + b0) @ W1 + b1)    dims 16 -> 16 -> 16
// one wave per 16 rows; 8 waves / block (256 thr).  K padded 16 -> 32.
// ---------------------------------------------------------------------------
__global__ void __launch_bounds__(256) lin_in_kernel(
    const float* __restrict__ x, const float* __restrict__ W0,
    const float* __restrict__ b0, const float* __restrict__ W1,
    const float* __restrict__ b1, float* __restrict__ h) {
  __shared__ alignas(32) _Float16 sW0[512];
  __shared__ alignas(32) _Float16 sW1[512];
  __shared__ float sb0[16], sb1[16];
  __shared__ alignas(32) _Float16 sA[8][16 * 32];
  __shared__ alignas(32) _Float16 sH[8][16 * 32];
  int tid = threadIdx.x;
  for (int i = tid; i < 512; i += 256) {
    int k = i >> 4, n = i & 15;
    sW0[swzB(k, n)] = (k < 16) ? (_Float16)W0[k * 16 + n] : (_Float16)0.f;
    sW1[swzB(k, n)] = (k < 16) ? (_Float16)W1[k * 16 + n] : (_Float16)0.f;
  }
  if (tid < 16) { sb0[tid] = b0[tid]; sb1[tid] = b1[tid]; }
  int w = tid >> 5, lane = tid & 31, col = lane & 15;
  long rowBase = (long)(blockIdx.x * 8 + w) * 16;
  for (int i = lane; i < 16 * 32; i += 32) {
    int r = i >> 5, f = i & 31;
    sA[w][i] = (f < 16) ? (_Float16)x[(rowBase + r) * 16 + f] : (_Float16)0.f;
  }
  __syncthreads();
  v16h a  = frag_A(sA[w], 32);
  v16h bw = frag_B_swz(sW0);
  v8f c; float bb = sb0[col];
#pragma unroll
  for (int r = 0; r < 8; ++r) c[r] = bb;
  c = wmma32(a, bw, c);
  int rg = (lane >> 4) * 8;
#pragma unroll
  for (int r = 0; r < 8; ++r) {
    sH[w][(rg + r) * 32 + col]      = (_Float16)fmaxf(c[r], 0.f);
    sH[w][(rg + r) * 32 + 16 + col] = (_Float16)0.f;   // zero K-pad
  }
  __syncthreads();
  v16h a2  = frag_A(sH[w], 32);
  v16h bw2 = frag_B_swz(sW1);
  v8f c2; float bb2 = sb1[col];
#pragma unroll
  for (int r = 0; r < 8; ++r) c2[r] = bb2;
  c2 = wmma32(a2, bw2, c2);
#pragma unroll
  for (int r = 0; r < 8; ++r)
    h[(rowBase + rg + r) * 16 + col] = fmaxf(c2[r], 0.f);
}

// ---------------------------------------------------------------------------
// FPS: sequential farthest point sampling, single block of 1024 threads.
// mind[] lives in d_ws (re-initialized every call -> deterministic).
// ---------------------------------------------------------------------------
__global__ void __launch_bounds__(1024) fps_kernel(
    const float* __restrict__ pos, int n, int m,
    int* __restrict__ outIdx, float* __restrict__ mind) {
  __shared__ float sval[1024];
  __shared__ int   sidx[1024];
  __shared__ int   sLast;
  __shared__ float sLx, sLy, sLz;
  int tid = threadIdx.x;
  for (int i = tid; i < n; i += 1024) mind[i] = 3.4e38f;
  if (tid == 0) { outIdx[0] = 0; sLast = 0; }
  __syncthreads();
  for (int s = 1; s < m; ++s) {
    if (tid == 0) {
      int l = sLast;
      sLx = pos[l * 3 + 0]; sLy = pos[l * 3 + 1]; sLz = pos[l * 3 + 2];
    }
    __syncthreads();
    float lx = sLx, ly = sLy, lz = sLz;
    float bestv = -1.f; int besti = 0;
    for (int i = tid; i < n; i += 1024) {
      if (i + 1024 < n) __builtin_prefetch(&pos[(i + 1024) * 3], 0, 1);
      float dx = pos[i * 3] - lx, dy = pos[i * 3 + 1] - ly,
            dz = pos[i * 3 + 2] - lz;
      float d = dx * dx + dy * dy + dz * dz;
      float mv = fminf(mind[i], d);
      mind[i] = mv;
      if (mv > bestv) { bestv = mv; besti = i; }
    }
    sval[tid] = bestv; sidx[tid] = besti;
    __syncthreads();
    for (int off = 512; off > 0; off >>= 1) {
      if (tid < off) {
        float ov = sval[tid + off]; int oi = sidx[tid + off];
        if (ov > sval[tid] || (ov == sval[tid] && oi < sidx[tid])) {
          sval[tid] = ov; sidx[tid] = oi;
        }
      }
      __syncthreads();
    }
    if (tid == 0) { outIdx[s] = sidx[0]; sLast = sidx[0]; }
    __syncthreads();
  }
}

__global__ void gather_pos_kernel(const float* __restrict__ pin,
                                  const int* __restrict__ idx,
                                  float* __restrict__ pout, int m) {
  int i = blockIdx.x * blockDim.x + threadIdx.x;
  if (i < m) {
    int s = idx[i];
    pout[i * 3 + 0] = pin[s * 3 + 0];
    pout[i * 3 + 1] = pin[s * 3 + 1];
    pout[i * 3 + 2] = pin[s * 3 + 2];
  }
}

// ---------------------------------------------------------------------------
// radius grouping: up to 32 nearest sources within r^2=4 per target,
// sorted ascending, cols = -1 for invalid slots.  one thread / target.
// ---------------------------------------------------------------------------
__global__ void __launch_bounds__(256) radius_group_kernel(
    const float* __restrict__ ptgt, int M, const float* __restrict__ psrc,
    int n, int* __restrict__ cols) {
  int t = blockIdx.x * blockDim.x + threadIdx.x;
  if (t >= M) return;
  float tx = ptgt[t * 3], ty = ptgt[t * 3 + 1], tz = ptgt[t * 3 + 2];
  float bd[32]; int bi[32]; int cnt = 0;
  for (int i = 0; i < n; ++i) {
    if ((i & 63) == 0 && i + 64 < n) __builtin_prefetch(&psrc[(i + 64) * 3], 0, 1);
    float dx = psrc[i * 3] - tx, dy = psrc[i * 3 + 1] - ty,
          dz = psrc[i * 3 + 2] - tz;
    float d = dx * dx + dy * dy + dz * dz;
    if (d <= 4.0f) {
      if (cnt < 32) {
        int j = cnt++;
        while (j > 0 && bd[j - 1] > d) { bd[j] = bd[j - 1]; bi[j] = bi[j - 1]; --j; }
        bd[j] = d; bi[j] = i;
      } else if (d < bd[31]) {
        int j = 31;
        while (j > 0 && bd[j - 1] > d) { bd[j] = bd[j - 1]; bi[j] = bi[j - 1]; --j; }
        bd[j] = d; bi[j] = i;
      }
    }
  }
  for (int j = 0; j < 32; ++j) cols[t * 32 + j] = (j < cnt) ? bi[j] : -1;
}

// ---------------------------------------------------------------------------
// SA: 4 targets / block (256 thr = 8 waves, 2 waves per target).
// per target: 32 messages [x_j(16) | rel_pos(3) | pad] -> 32x32 f16 tile,
// MLP 19->19 (relu) ->16 (relu) via WMMA, invalid rows -> -1e30, column max.
// ---------------------------------------------------------------------------
__global__ void __launch_bounds__(256) sa_kernel(
    const float* __restrict__ xsrc, const float* __restrict__ psrc,
    const float* __restrict__ ptgt, const int* __restrict__ cols,
    const float* __restrict__ W0, const float* __restrict__ b0,
    const float* __restrict__ W1, const float* __restrict__ b1,
    float* __restrict__ xout) {
  __shared__ alignas(32) _Float16 sW0[2 * 512];   // two swizzled N-tiles
  __shared__ alignas(32) _Float16 sW1[512];
  __shared__ float sb0[32], sb1[16];
  __shared__ alignas(32) _Float16 sA[4][32 * 32];
  __shared__ alignas(32) _Float16 sH[4][32 * 32];
  __shared__ float sOut[4][64];
  __shared__ int sC[4][32];
  int tid = threadIdx.x;
  // weights, swizzled, shared by all 4 targets
  for (int i = tid; i < 1024; i += 256) {
    int k = i >> 5, n = i & 31;
    float v = (k < 19 && n < 19) ? W0[k * 19 + n] : 0.f;
    sW0[(n >> 4) * 512 + swzB(k, n & 15)] = (_Float16)v;
  }
  for (int i = tid; i < 512; i += 256) {
    int k = i >> 4, n = i & 15;
    sW1[swzB(k, n)] = (k < 19) ? (_Float16)W1[k * 16 + n] : (_Float16)0.f;
  }
  if (tid < 32) sb0[tid] = (tid < 19) ? b0[tid] : 0.f;
  if (tid < 16) sb1[tid] = b1[tid];
  int s = tid >> 6;            // sub-block (target slot) 0..3
  int stid = tid & 63;
  int t = blockIdx.x * 4 + s;
  if (stid < 32) sC[s][stid] = cols[t * 32 + stid];
  __syncthreads();
  float tx = ptgt[t * 3], ty = ptgt[t * 3 + 1], tz = ptgt[t * 3 + 2];
  // build A: 32 msgs x 32 feats per target
  for (int i = stid; i < 32 * 32; i += 64) {
    int j = i >> 5, f = i & 31;
    int c = sC[s][j];
    float v = 0.f;
    if (c >= 0) {
      if (f < 16)       v = xsrc[c * 16 + f];
      else if (f == 16) v = psrc[c * 3 + 0] - tx;
      else if (f == 17) v = psrc[c * 3 + 1] - ty;
      else if (f == 18) v = psrc[c * 3 + 2] - tz;
    }
    sA[s][i] = (_Float16)v;
  }
  __syncthreads();
  int g = (stid >> 5);         // wave within target (rows g*16 .. g*16+15)
  int lane = tid & 31, col = lane & 15;
  int rg = (lane >> 4) * 8;
  // layer 0: 19(pad32) -> 19(pad32), two N-tiles
  v16h a  = frag_A(sA[s] + g * 16 * 32, 32);
  v16h w0 = frag_B_swz(sW0);
  v16h w1 = frag_B_swz(sW0 + 512);
  v8f c0, c1; float bb0 = sb0[col], bb1 = sb0[16 + col];
#pragma unroll
  for (int r = 0; r < 8; ++r) { c0[r] = bb0; c1[r] = bb1; }
  c0 = wmma32(a, w0, c0);
  c1 = wmma32(a, w1, c1);
#pragma unroll
  for (int r = 0; r < 8; ++r) {
    int row = g * 16 + rg + r;
    sH[s][row * 32 + col]      = (_Float16)fmaxf(c0[r], 0.f);
    sH[s][row * 32 + 16 + col] = (_Float16)fmaxf(c1[r], 0.f);
  }
  __syncthreads();
  // layer 1: 19(pad32) -> 16, then per-lane masked max over its 8 rows
  v16h a2 = frag_A(sH[s] + g * 16 * 32, 32);
  v16h w2 = frag_B_swz(sW1);
  v8f c2; float bb2 = sb1[col];
#pragma unroll
  for (int r = 0; r < 8; ++r) c2[r] = bb2;
  c2 = wmma32(a2, w2, c2);
  float lmax = -3.4e38f;
#pragma unroll
  for (int r = 0; r < 8; ++r) {
    int row = g * 16 + rg + r;
    float v = fmaxf(c2[r], 0.f);
    if (sC[s][row] < 0) v = -1e30f;
    lmax = fmaxf(lmax, v);
  }
  sOut[s][g * 32 + lane] = lmax;   // lane = (half<<4)|col
  __syncthreads();
  if (stid < 16) {
    float mx = fmaxf(fmaxf(sOut[s][stid], sOut[s][16 + stid]),
                     fmaxf(sOut[s][32 + stid], sOut[s][48 + stid]));
    xout[t * 16 + stid] = mx;
  }
}

// ---------------------------------------------------------------------------
// kNN (k=3) inverse-distance interpolation.  one thread / fine point.
// ---------------------------------------------------------------------------
__global__ void __launch_bounds__(256) knn_interp_kernel(
    const float* __restrict__ pf, int Nf, const float* __restrict__ pc,
    int Nc, const float* __restrict__ xc, float* __restrict__ xi) {
  int i = blockIdx.x * blockDim.x + threadIdx.x;
  if (i >= Nf) return;
  float fx = pf[i * 3], fy = pf[i * 3 + 1], fz = pf[i * 3 + 2];
  float d0 = 3.4e38f, d1 = 3.4e38f, d2 = 3.4e38f;
  int i0 = 0, i1 = 0, i2 = 0;
  for (int j = 0; j < Nc; ++j) {
    if ((j & 63) == 0 && j + 64 < Nc) __builtin_prefetch(&pc[(j + 64) * 3], 0, 1);
    float dx = pc[j * 3] - fx, dy = pc[j * 3 + 1] - fy, dz = pc[j * 3 + 2] - fz;
    float d = dx * dx + dy * dy + dz * dz;
    if (d < d0)      { d2 = d1; i2 = i1; d1 = d0; i1 = i0; d0 = d; i0 = j; }
    else if (d < d1) { d2 = d1; i2 = i1; d1 = d; i1 = j; }
    else if (d < d2) { d2 = d; i2 = j; }
  }
  float w0 = 1.f / fmaxf(d0, 1e-16f);
  float w1 = 1.f / fmaxf(d1, 1e-16f);
  float w2 = 1.f / fmaxf(d2, 1e-16f);
  float ws = w0 + w1 + w2;
  w0 /= ws; w1 /= ws; w2 /= ws;
  for (int c = 0; c < 16; ++c)
    xi[i * 16 + c] =
        w0 * xc[i0 * 16 + c] + w1 * xc[i1 * 16 + c] + w2 * xc[i2 * 16 + c];
}

// ---------------------------------------------------------------------------
// FP MLP: xf = relu(relu([xi|xs] @ W0 + b0) @ W1 + b1)   dims 32 -> 32 -> 16
// ---------------------------------------------------------------------------
__global__ void __launch_bounds__(256) fp_mlp_kernel(
    const float* __restrict__ xi, const float* __restrict__ xs,
    const float* __restrict__ W0, const float* __restrict__ b0,
    const float* __restrict__ W1, const float* __restrict__ b1,
    float* __restrict__ xf) {
  __shared__ alignas(32) _Float16 sW0[2 * 512];
  __shared__ alignas(32) _Float16 sW1[512];
  __shared__ float sb0[32], sb1[16];
  __shared__ alignas(32) _Float16 sA[8][16 * 32];
  __shared__ alignas(32) _Float16 sH[8][16 * 32];
  int tid = threadIdx.x;
  for (int i = tid; i < 1024; i += 256) {
    int k = i >> 5, n = i & 31;
    sW0[(n >> 4) * 512 + swzB(k, n & 15)] = (_Float16)W0[k * 32 + n];
  }
  for (int i = tid; i < 512; i += 256) {
    int k = i >> 4, n = i & 15;
    sW1[swzB(k, n)] = (_Float16)W1[k * 16 + n];
  }
  if (tid < 32) sb0[tid] = b0[tid];
  if (tid < 16) sb1[tid] = b1[tid];
  int w = tid >> 5, lane = tid & 31, col = lane & 15;
  long rowBase = (long)(blockIdx.x * 8 + w) * 16;
  for (int i = lane; i < 16 * 32; i += 32) {
    int r = i >> 5, f = i & 31;
    long gi = rowBase + r;
    float v = (f < 16) ? xi[gi * 16 + f] : xs[gi * 16 + (f - 16)];
    sA[w][i] = (_Float16)v;
  }
  __syncthreads();
  v16h a  = frag_A(sA[w], 32);
  v16h w0 = frag_B_swz(sW0);
  v16h w1 = frag_B_swz(sW0 + 512);
  v8f c0, c1; float bb0 = sb0[col], bb1 = sb0[16 + col];
#pragma unroll
  for (int r = 0; r < 8; ++r) { c0[r] = bb0; c1[r] = bb1; }
  c0 = wmma32(a, w0, c0);
  c1 = wmma32(a, w1, c1);
  int rg = (lane >> 4) * 8;
#pragma unroll
  for (int r = 0; r < 8; ++r) {
    sH[w][(rg + r) * 32 + col]      = (_Float16)fmaxf(c0[r], 0.f);
    sH[w][(rg + r) * 32 + 16 + col] = (_Float16)fmaxf(c1[r], 0.f);
  }
  __syncthreads();
  v16h a2 = frag_A(sH[w], 32);
  v16h w2 = frag_B_swz(sW1);
  v8f c2; float bb2 = sb1[col];
#pragma unroll
  for (int r = 0; r < 8; ++r) c2[r] = bb2;
  c2 = wmma32(a2, w2, c2);
#pragma unroll
  for (int r = 0; r < 8; ++r)
    xf[(rowBase + rg + r) * 16 + col] = fmaxf(c2[r], 0.f);
}

// ---------------------------------------------------------------------------
// lin_out:  out = relu(xf @ W0 + b0) @ W1 + b1     dims 16 -> 16 -> 2
// ---------------------------------------------------------------------------
__global__ void __launch_bounds__(256) lin_out_kernel(
    const float* __restrict__ xf, const float* __restrict__ W0,
    const float* __restrict__ b0, const float* __restrict__ W1,
    const float* __restrict__ b1, float* __restrict__ out) {
  __shared__ alignas(32) _Float16 sW0[512];
  __shared__ alignas(32) _Float16 sW1[512];
  __shared__ float sb0[16], sb1[16];
  __shared__ alignas(32) _Float16 sA[8][16 * 32];
  __shared__ alignas(32) _Float16 sH[8][16 * 32];
  int tid = threadIdx.x;
  for (int i = tid; i < 512; i += 256) {
    int k = i >> 4, n = i & 15;
    sW0[swzB(k, n)] = (k < 16) ? (_Float16)W0[k * 16 + n] : (_Float16)0.f;
    sW1[swzB(k, n)] =
        (k < 16 && n < 2) ? (_Float16)W1[k * 2 + n] : (_Float16)0.f;
  }
  if (tid < 16) {
    sb0[tid] = b0[tid];
    sb1[tid] = (tid < 2) ? b1[tid] : 0.f;
  }
  int w = tid >> 5, lane = tid & 31, col = lane & 15;
  long rowBase = (long)(blockIdx.x * 8 + w) * 16;
  for (int i = lane; i < 16 * 32; i += 32) {
    int r = i >> 5, f = i & 31;
    sA[w][i] = (f < 16) ? (_Float16)xf[(rowBase + r) * 16 + f] : (_Float16)0.f;
  }
  __syncthreads();
  v16h a  = frag_A(sA[w], 32);
  v16h w0 = frag_B_swz(sW0);
  v8f c0; float bb0 = sb0[col];
#pragma unroll
  for (int r = 0; r < 8; ++r) c0[r] = bb0;
  c0 = wmma32(a, w0, c0);
  int rg = (lane >> 4) * 8;
#pragma unroll
  for (int r = 0; r < 8; ++r) {
    sH[w][(rg + r) * 32 + col]      = (_Float16)fmaxf(c0[r], 0.f);
    sH[w][(rg + r) * 32 + 16 + col] = (_Float16)0.f;
  }
  __syncthreads();
  v16h a2 = frag_A(sH[w], 32);
  v16h w1 = frag_B_swz(sW1);
  v8f c2; float bb2 = sb1[col];
#pragma unroll
  for (int r = 0; r < 8; ++r) c2[r] = bb2;
  c2 = wmma32(a2, w1, c2);          // final layer: NO relu
  if (col < 2) {
#pragma unroll
    for (int r = 0; r < 8; ++r)
      out[(rowBase + rg + r) * 2 + col] = c2[r];
  }
}

// ---------------------------------------------------------------------------
// host orchestration
// ---------------------------------------------------------------------------
extern "C" void kernel_launch(void* const* d_in, const int* in_sizes, int n_in,
                              void* d_out, int out_size, void* d_ws,
                              size_t ws_size, hipStream_t stream) {
  (void)in_sizes; (void)n_in; (void)out_size; (void)ws_size;
  const int N = 16384, M0 = 8192, M1 = 4096, M2 = 2048;

  const float* x    = (const float*)d_in[0];
  const float* pos  = (const float*)d_in[1];
  // d_in[2] = batch (all zeros) -- unused
  const float* liW0 = (const float*)d_in[3];
  const float* lib0 = (const float*)d_in[4];
  const float* liW1 = (const float*)d_in[5];
  const float* lib1 = (const float*)d_in[6];
  const float* saW0 = (const float*)d_in[7];   // [3,19,19]
  const float* sab0 = (const float*)d_in[8];   // [3,19]
  const float* saW1 = (const float*)d_in[9];   // [3,19,16]
  const float* sab1 = (const float*)d_in[10];  // [3,16]
  const float* fpW0 = (const float*)d_in[11];  // [3,32,32]
  const float* fpb0 = (const float*)d_in[12];  // [3,32]
  const float* fpW1 = (const float*)d_in[13];  // [3,32,16]
  const float* fpb1 = (const float*)d_in[14];  // [3,16]
  const float* loW0 = (const float*)d_in[15];
  const float* lob0 = (const float*)d_in[16];
  const float* loW1 = (const float*)d_in[17];  // [16,2]
  const float* lob1 = (const float*)d_in[18];
  float* out = (float*)d_out;

  size_t off = 0;
  auto ALLOC = [&](size_t nbytes) -> void* {
    void* p = (void*)((char*)d_ws + off);
    off += (nbytes + 255) & ~(size_t)255;
    return p;
  };
  float* h0   = (float*)ALLOC((size_t)N  * 16 * 4);
  float* mind = (float*)ALLOC((size_t)N  * 4);
  int*   idx0 = (int*)  ALLOC((size_t)M0 * 4);
  int*   idx1 = (int*)  ALLOC((size_t)M1 * 4);
  int*   idx2 = (int*)  ALLOC((size_t)M2 * 4);
  float* pos1 = (float*)ALLOC((size_t)M0 * 3 * 4);
  float* pos2 = (float*)ALLOC((size_t)M1 * 3 * 4);
  float* pos3 = (float*)ALLOC((size_t)M2 * 3 * 4);
  int*   cols = (int*)  ALLOC((size_t)M0 * 32 * 4);
  float* x1   = (float*)ALLOC((size_t)M0 * 16 * 4);
  float* x2   = (float*)ALLOC((size_t)M1 * 16 * 4);
  float* x3   = (float*)ALLOC((size_t)M2 * 16 * 4);
  float* xi   = (float*)ALLOC((size_t)N  * 16 * 4);
  float* xf2  = (float*)ALLOC((size_t)M1 * 16 * 4);
  float* xf1  = (float*)ALLOC((size_t)M0 * 16 * 4);
  float* xf0  = (float*)ALLOC((size_t)N  * 16 * 4);

  // 1. lin_in
  lin_in_kernel<<<N / 128, 256, 0, stream>>>(x, liW0, lib0, liW1, lib1, h0);

  // 2. SA level 0: 16384 -> 8192
  fps_kernel<<<1, 1024, 0, stream>>>(pos, N, M0, idx0, mind);
  gather_pos_kernel<<<(M0 + 255) / 256, 256, 0, stream>>>(pos, idx0, pos1, M0);
  radius_group_kernel<<<(M0 + 255) / 256, 256, 0, stream>>>(pos1, M0, pos, N, cols);
  sa_kernel<<<M0 / 4, 256, 0, stream>>>(h0, pos, pos1, cols,
                                        saW0 + 0 * 19 * 19, sab0 + 0 * 19,
                                        saW1 + 0 * 19 * 16, sab1 + 0 * 16, x1);

  // 3. SA level 1: 8192 -> 4096
  fps_kernel<<<1, 1024, 0, stream>>>(pos1, M0, M1, idx1, mind);
  gather_pos_kernel<<<(M1 + 255) / 256, 256, 0, stream>>>(pos1, idx1, pos2, M1);
  radius_group_kernel<<<(M1 + 255) / 256, 256, 0, stream>>>(pos2, M1, pos1, M0, cols);
  sa_kernel<<<M1 / 4, 256, 0, stream>>>(x1, pos1, pos2, cols,
                                        saW0 + 1 * 19 * 19, sab0 + 1 * 19,
                                        saW1 + 1 * 19 * 16, sab1 + 1 * 16, x2);

  // 4. SA level 2: 4096 -> 2048
  fps_kernel<<<1, 1024, 0, stream>>>(pos2, M1, M2, idx2, mind);
  gather_pos_kernel<<<(M2 + 255) / 256, 256, 0, stream>>>(pos2, idx2, pos3, M2);
  radius_group_kernel<<<(M2 + 255) / 256, 256, 0, stream>>>(pos3, M2, pos2, M1, cols);
  sa_kernel<<<M2 / 4, 256, 0, stream>>>(x2, pos2, pos3, cols,
                                        saW0 + 2 * 19 * 19, sab0 + 2 * 19,
                                        saW1 + 2 * 19 * 16, sab1 + 2 * 16, x3);

  // 5. FP level 2: (x3,pos3) -> pos2, skip x2
  knn_interp_kernel<<<(M1 + 255) / 256, 256, 0, stream>>>(pos2, M1, pos3, M2, x3, xi);
  fp_mlp_kernel<<<M1 / 128, 256, 0, stream>>>(xi, x2,
                                              fpW0 + 2 * 32 * 32, fpb0 + 2 * 32,
                                              fpW1 + 2 * 32 * 16, fpb1 + 2 * 16, xf2);

  // 6. FP level 1: (xf2,pos2) -> pos1, skip x1
  knn_interp_kernel<<<(M0 + 255) / 256, 256, 0, stream>>>(pos1, M0, pos2, M1, xf2, xi);
  fp_mlp_kernel<<<M0 / 128, 256, 0, stream>>>(xi, x1,
                                              fpW0 + 1 * 32 * 32, fpb0 + 1 * 32,
                                              fpW1 + 1 * 32 * 16, fpb1 + 1 * 16, xf1);

  // 7. FP level 0: (xf1,pos1) -> pos, skip h0
  knn_interp_kernel<<<(N + 255) / 256, 256, 0, stream>>>(pos, N, pos1, M0, xf1, xi);
  fp_mlp_kernel<<<N / 128, 256, 0, stream>>>(xi, h0,
                                             fpW0 + 0 * 32 * 32, fpb0 + 0 * 32,
                                             fpW1 + 0 * 32 * 16, fpb1 + 0 * 16, xf0);

  // 8. lin_out -> [N,2]
  lin_out_kernel<<<N / 128, 256, 0, stream>>>(xf0, loW0, lob0, loW1, lob1, out);
}
